// Experts_37718402793888
// MI455X (gfx1250) — compile-verified, matrix operand order checked by reading, not run on hardware
//
#include <hip/hip_runtime.h>
#include <hip/hip_bf16.h>

// Problem constants (match reference)
#define NEXP    8
#define D_MODEL 2048
#define D_FF    8192
#define CAP     1024

// Tile config: 128x128 block tile, K-step 32 (WMMA bf16 K), 256 threads = 8 wave32
#define BM   128
#define BN   128
#define BK   32
#define LDSK 40   // padded LDS row stride (elements): 80B = 20 banks -> conflict-free b128 reads

#if defined(__gfx1250__)
#define USE_TDM 1
#else
#define USE_TDM 0
#endif

typedef __attribute__((ext_vector_type(16))) __bf16 v16bf;
typedef __attribute__((ext_vector_type(8)))  __bf16 v8bf;
typedef __attribute__((ext_vector_type(2)))  __bf16 v2bf;
typedef __attribute__((ext_vector_type(8)))  float  v8f;
typedef __attribute__((ext_vector_type(4)))  unsigned int v4u;
typedef __attribute__((ext_vector_type(8)))  unsigned int v8u;

__device__ __forceinline__ unsigned short f2bf_bits(float f) {
  union { float f; unsigned u; } c; c.f = f;
  unsigned r = c.u + 0x7FFFu + ((c.u >> 16) & 1u);   // round-to-nearest-even
  return (unsigned short)(r >> 16);
}

// Pack two f32 -> two bf16 in one dword. Prefer hardware v_cvt_pk_bf16_f32.
#if __has_builtin(__builtin_amdgcn_cvt_pk_bf16_f32)
__device__ __forceinline__ unsigned pk_bf16(float a, float b) {
  v2bf p = __builtin_amdgcn_cvt_pk_bf16_f32(a, b);
  unsigned u; __builtin_memcpy(&u, &p, 4);
  return u;
}
#else
__device__ __forceinline__ unsigned pk_bf16(float a, float b) {
  return (unsigned)f2bf_bits(a) | ((unsigned)f2bf_bits(b) << 16);
}
#endif

// Branch-free tanh-approx GELU: native v_exp_f32, no libm slow path / EXEC churn.
__device__ __forceinline__ float gelu_tanh(float x) {
  const float k0 = 0.7978845608028654f;   // sqrt(2/pi)
  const float k1 = 0.044715f;
  float t  = k0 * (x + k1 * x * x * x);
  float at = fabsf(t);
  float e  = __expf(-2.0f * at);                 // v_exp_f32
  float th = __fdividef(1.0f - e, 1.0f + e);     // fast rcp path
  th = copysignf(th, t);
  return 0.5f * x * (1.0f + th);
}

__device__ __forceinline__ void wait_tensorcnt0() {
#if __has_builtin(__builtin_amdgcn_s_wait_tensorcnt)
  __builtin_amdgcn_s_wait_tensorcnt(0);
#else
  asm volatile("s_wait_tensorcnt 0x0" ::: "memory");
#endif
}

// Load one 16x32 bf16 WMMA A/B fragment from padded LDS tile (row-major [rows][LDSK]).
// ISA 16-bit A layout: lanes 0-15 -> M=lane, K=0..7 (V0..3) & K=16..23 (V4..7);
//                      lanes 16-31 -> M=lane-16, K=8..15 & K=24..31.
__device__ __forceinline__ v16bf load_frag(const unsigned short* lds, int rowBase, int lane) {
  const int row = rowBase + (lane & 15);
  const int hk  = (lane >> 4) * 8;
  const v8bf lo = *(const v8bf*)&lds[row * LDSK + hk];        // K = hk .. hk+7
  const v8bf hi = *(const v8bf*)&lds[row * LDSK + hk + 16];   // K = 16+hk .. 16+hk+7
  v16bf r;
#pragma unroll
  for (int i = 0; i < 8; ++i) { r[i] = lo[i]; r[i + 8] = hi[i]; }
  return r;
}

// ---------------- GEMM1: H = gelu(X @ W1 + b1), bf16 output ----------------
__global__ __launch_bounds__(256)
void ffn_gemm1_gelu(const float* __restrict__ X,    // [E, C, D]
                    const float* __restrict__ W1,   // [E, D, F]
                    const float* __restrict__ b1,   // [E, F]
                    unsigned short* __restrict__ H) // bf16 [E, C, F]
{
  __shared__ unsigned short ldsA[BM * LDSK];
  __shared__ unsigned short ldsB[BN * LDSK];

  const int fTile = blockIdx.x * BN;
  const int cTile = blockIdx.y * BM;
  const int e     = blockIdx.z;

  const int tid   = threadIdx.x;
  const int lane  = tid & 31;
  const int wave  = tid >> 5;
  const int waveM = (wave & 3) * 32;   // 4 waves along M
  const int waveN = (wave >> 2) * 64;  // 2 waves along N

  const float* Xe  = X  + (size_t)e * CAP * D_MODEL + (size_t)cTile * D_MODEL;
  const float* W1e = W1 + (size_t)e * D_MODEL * (size_t)D_FF + fTile;
  const float* b1e = b1 + (size_t)e * D_FF + fTile;

  v8f acc[2][4];
#pragma unroll
  for (int mi = 0; mi < 2; ++mi)
#pragma unroll
    for (int ni = 0; ni < 4; ++ni)
      acc[mi][ni] = (v8f)(0.0f);

  for (int k0 = 0; k0 < D_MODEL; k0 += BK) {
    // Stage A tile: 128x32 f32 -> bf16 (row-major). 1024 float4 / 256 threads,
    // packed pairs -> one ds_store_b64 per float4.
#pragma unroll
    for (int i = 0; i < 4; ++i) {
      const int idx = tid + i * 256;       // float4 index
      const int row = idx >> 3;            // 8 float4 per row of 32
      const int c4  = (idx & 7) * 4;
      const float4 v = *(const float4*)&Xe[(size_t)row * D_MODEL + k0 + c4];
      uint2 s;
      s.x = pk_bf16(v.x, v.y);
      s.y = pk_bf16(v.z, v.w);
      *(uint2*)&ldsA[row * LDSK + c4] = s;
    }
    // Stage B tile transposed: W1 rows k0..k0+31, cols fTile..+127 -> ldsB[n][k].
    // Load two consecutive k-rows, pack (k,k+1) pairs -> ds_store_b32 per column.
#pragma unroll
    for (int i = 0; i < 2; ++i) {
      const int p  = tid + i * 256;        // pair-work index, 0..511
      const int kp = (p >> 5) * 2;         // even k row (0..30)
      const int f4 = (p & 31) * 4;
      const float4 a = *(const float4*)&W1e[(size_t)(k0 + kp) * D_FF + f4];
      const float4 b = *(const float4*)&W1e[(size_t)(k0 + kp + 1) * D_FF + f4];
      *(unsigned*)&ldsB[(f4 + 0) * LDSK + kp] = pk_bf16(a.x, b.x);
      *(unsigned*)&ldsB[(f4 + 1) * LDSK + kp] = pk_bf16(a.y, b.y);
      *(unsigned*)&ldsB[(f4 + 2) * LDSK + kp] = pk_bf16(a.z, b.z);
      *(unsigned*)&ldsB[(f4 + 3) * LDSK + kp] = pk_bf16(a.w, b.w);
    }
    // Prefetch next k-tile (global_prefetch_b8) while this stage computes.
    if (k0 + BK < D_MODEL) {
      __builtin_prefetch(&Xe[(size_t)(tid >> 1) * D_MODEL + k0 + BK + (tid & 1) * 16], 0, 1);
      __builtin_prefetch(&W1e[(size_t)(k0 + BK + (tid >> 3)) * D_FF + (tid & 7) * 16], 0, 1);
    }
    __syncthreads();

    v16bf fa[2], fb[4];
#pragma unroll
    for (int mi = 0; mi < 2; ++mi) fa[mi] = load_frag(ldsA, waveM + mi * 16, lane);
#pragma unroll
    for (int ni = 0; ni < 4; ++ni) fb[ni] = load_frag(ldsB, waveN + ni * 16, lane);

#pragma unroll
    for (int mi = 0; mi < 2; ++mi)
#pragma unroll
      for (int ni = 0; ni < 4; ++ni)
        acc[mi][ni] = __builtin_amdgcn_wmma_f32_16x16x32_bf16(
            false, fa[mi], false, fb[ni], (short)0, acc[mi][ni], false, false);
    __syncthreads();
  }

  // Epilogue: bias + gelu -> bf16 H. C/D layout: lane%16 = N col; VGPR r -> M = r + 8*(lane>>4).
  const int colBase = lane & 15;
  const int rowHalf = (lane >> 4) * 8;
#pragma unroll
  for (int mi = 0; mi < 2; ++mi) {
#pragma unroll
    for (int ni = 0; ni < 4; ++ni) {
      const int col  = waveN + ni * 16 + colBase;
      const float bb = b1e[col];
#pragma unroll
      for (int r = 0; r < 8; ++r) {
        const int row = waveM + mi * 16 + rowHalf + r;
        const float y = gelu_tanh(acc[mi][ni][r] + bb);
        const size_t o = ((size_t)e * CAP + (size_t)(cTile + row)) * D_FF + (fTile + col);
        H[o] = f2bf_bits(y);
      }
    }
  }
}

// ---------------- GEMM2: Y = H @ W2 + b2, f32 output ----------------
__global__ __launch_bounds__(256)
void ffn_gemm2(const unsigned short* __restrict__ H, // bf16 [E, C, F]
               const float* __restrict__ W2,         // [E, F, D]
               const float* __restrict__ b2,         // [E, D]
               float* __restrict__ Y)                // [E, C, D]
{
  __shared__ unsigned short ldsA[BM * LDSK];
  __shared__ unsigned short ldsB[BN * LDSK];

  const int dTile = blockIdx.x * BN;
  const int cTile = blockIdx.y * BM;
  const int e     = blockIdx.z;

  const int tid   = threadIdx.x;
  const int lane  = tid & 31;
  const int wave  = tid >> 5;
  const int waveM = (wave & 3) * 32;
  const int waveN = (wave >> 2) * 64;

  const unsigned short* He = H + ((size_t)e * CAP + (size_t)cTile) * D_FF;
  const float* W2e = W2 + (size_t)e * (size_t)D_FF * D_MODEL + dTile;
  const float* b2e = b2 + (size_t)e * D_MODEL + dTile;

  v8f acc[2][4];
#pragma unroll
  for (int mi = 0; mi < 2; ++mi)
#pragma unroll
    for (int ni = 0; ni < 4; ++ni)
      acc[mi][ni] = (v8f)(0.0f);

  for (int k0 = 0; k0 < D_FF; k0 += BK) {
#if USE_TDM
    // --- Tensor Data Mover stages the whole 128x32 bf16 A tile into LDS. ---
    // D# pad fields reproduce the padded LDS layout: 16 DWORDs (64B) of row data
    // then 4 DWORDs (16B) pad -> 80B row stride == LDSK bf16 elements.
    if (wave == 0) {
      const unsigned long long gaddr =
          (unsigned long long)(uintptr_t)He + (unsigned long long)k0 * 2ull;
      const unsigned ldsOff = (unsigned)(uintptr_t)&ldsA[0];   // generic ptr low 32b = LDS offset
      const unsigned TD0  = D_FF / 2;   // tensor dim0 length in DWORDs (row of H)
      const unsigned TD1  = CAP;        // tensor dim1 (rows)
      const unsigned TD0S = D_FF / 2;   // dim0 stride in DWORDs
      v4u g0;
      g0[0] = 1u;                                     // count=1 valid user descriptor
      g0[1] = ldsOff;                                 // lds_addr (bytes)
      g0[2] = (unsigned)(gaddr & 0xFFFFFFFFu);        // global_addr[31:0]
      g0[3] = (unsigned)((gaddr >> 32) & 0x01FFFFFFu) // global_addr[56:32]
            | (2u << 30);                             // type = 2 ("image")
      v8u g1;
      g1[0] = (2u << 16)        // data_size = 4 bytes
            | (1u << 20)        // pad_enable
            | (3u << 22)        // pad_interval: 16 DWORDs of data
            | (3u << 25);       // pad_amount:   4 DWORDs of pad
      g1[1] = (TD0 & 0xFFFFu) << 16;                  // tensor_dim0 lo16 (atomic_barrier=0)
      g1[2] = (TD0 >> 16) | ((TD1 & 0xFFFFu) << 16);  // tensor_dim0 hi16 | tensor_dim1 lo16
      g1[3] = (TD1 >> 16) | (16u << 16);              // tensor_dim1 hi16 | tile_dim0 = 16 DWORDs
      g1[4] = 128u;                                   // tile_dim1 = 128 rows, tile_dim2 = 0
      g1[5] = TD0S;                                   // tensor_dim0_stride lo32
      g1[6] = 0u;                                     // stride0 hi16 | dim1_stride lo16
      g1[7] = 0u;
      asm volatile("tensor_load_to_lds %0, %1" :: "s"(g0), "s"(g1) : "memory");
    }
#else
    // Fallback: manual bf16 copy, 128x32 elems = 512 16B chunks / 256 threads.
#pragma unroll
    for (int i = 0; i < 2; ++i) {
      const int idx = tid + i * 256;
      const int row = idx >> 2;            // 4 chunks per row of 32
      const int c8  = (idx & 3) * 8;
      const uint4 v = *(const uint4*)&He[(size_t)row * D_FF + k0 + c8];
      *(uint4*)&ldsA[row * LDSK + c8] = v;
    }
#endif
    // Stage B tile transposed: W2 rows k0..k0+31, cols dTile..+127 -> ldsB[n][k].
#pragma unroll
    for (int i = 0; i < 2; ++i) {
      const int p  = tid + i * 256;        // pair-work index, 0..511
      const int kp = (p >> 5) * 2;         // even k row (0..30)
      const int d4 = (p & 31) * 4;
      const float4 a = *(const float4*)&W2e[(size_t)(k0 + kp) * D_MODEL + d4];
      const float4 b = *(const float4*)&W2e[(size_t)(k0 + kp + 1) * D_MODEL + d4];
      *(unsigned*)&ldsB[(d4 + 0) * LDSK + kp] = pk_bf16(a.x, b.x);
      *(unsigned*)&ldsB[(d4 + 1) * LDSK + kp] = pk_bf16(a.y, b.y);
      *(unsigned*)&ldsB[(d4 + 2) * LDSK + kp] = pk_bf16(a.z, b.z);
      *(unsigned*)&ldsB[(d4 + 3) * LDSK + kp] = pk_bf16(a.w, b.w);
    }
    // Prefetch next W2 k-tile.
    if (k0 + BK < D_FF) {
      __builtin_prefetch(&W2e[(size_t)(k0 + BK + (tid >> 3)) * D_MODEL + (tid & 7) * 16], 0, 1);
    }
#if USE_TDM
    if (wave == 0) wait_tensorcnt0();
#endif
    __syncthreads();

    v16bf fa[2], fb[4];
#pragma unroll
    for (int mi = 0; mi < 2; ++mi) fa[mi] = load_frag(ldsA, waveM + mi * 16, lane);
#pragma unroll
    for (int ni = 0; ni < 4; ++ni) fb[ni] = load_frag(ldsB, waveN + ni * 16, lane);

#pragma unroll
    for (int mi = 0; mi < 2; ++mi)
#pragma unroll
      for (int ni = 0; ni < 4; ++ni)
        acc[mi][ni] = __builtin_amdgcn_wmma_f32_16x16x32_bf16(
            false, fa[mi], false, fb[ni], (short)0, acc[mi][ni], false, false);
    __syncthreads();
  }

  const int colBase = lane & 15;
  const int rowHalf = (lane >> 4) * 8;
#pragma unroll
  for (int mi = 0; mi < 2; ++mi) {
#pragma unroll
    for (int ni = 0; ni < 4; ++ni) {
      const int col  = waveN + ni * 16 + colBase;
      const float bb = b2e[col];
#pragma unroll
      for (int r = 0; r < 8; ++r) {
        const int row = waveM + mi * 16 + rowHalf + r;
        const size_t o = ((size_t)e * CAP + (size_t)(cTile + row)) * D_MODEL + (dTile + col);
        Y[o] = acc[mi][ni][r] + bb;
      }
    }
  }
}

extern "C" void kernel_launch(void* const* d_in, const int* in_sizes, int n_in,
                              void* d_out, int out_size, void* d_ws, size_t ws_size,
                              hipStream_t stream) {
  (void)in_sizes; (void)n_in; (void)out_size; (void)ws_size;
  const float* X  = (const float*)d_in[0];   // [B=1, E*C, D]
  const float* W1 = (const float*)d_in[1];   // [E, D, F]
  const float* b1 = (const float*)d_in[2];   // [E, F]
  const float* W2 = (const float*)d_in[3];   // [E, F, D]
  const float* b2 = (const float*)d_in[4];   // [E, D]
  float* out = (float*)d_out;                // [B=1, E*C, D] f32

  // Workspace: H intermediate as bf16 [E, C, F] = 128 MB
  unsigned short* H = (unsigned short*)d_ws;

  dim3 blk(256, 1, 1);
  dim3 g1(D_FF / BN, CAP / BM, NEXP);     // (64, 8, 8)
  ffn_gemm1_gelu<<<g1, blk, 0, stream>>>(X, W1, b1, H);

  dim3 g2(D_MODEL / BN, CAP / BM, NEXP);  // (16, 8, 8)
  ffn_gemm2<<<g2, blk, 0, stream>>>(H, W2, b2, out);
}
